// DeeperGCN_41188736369374
// MI455X (gfx1250) — matrix operand order repaired and checked by reading, not compiled, not run on hardware
//
#include <hip/hip_runtime.h>

// ---------------- constants (from reference) ----------------
#define MSG_EPS 1e-7f

// ---------------- WMMA plumbing (gfx1250, wave32) ----------------
typedef __attribute__((ext_vector_type(16))) __bf16 v16bf;
typedef __attribute__((ext_vector_type(8)))  float  v8f;
typedef __attribute__((ext_vector_type(4)))  float  f32x4;   // for NT vector lds

union ABFrag { v16bf v; unsigned w[8]; };     // built from f32 via pk converts
union BFrag  { v16bf v; uint4 q[2]; };        // loaded pre-packed from memory

// Pack two f32 into a (lo,hi) bf16 pair. CDNA5 VOP3P ops 62/63:
// V_FMA_MIXLO_BF16 / V_FMA_MIXHI_BF16: D[15:0]/[31:16].bf16 = cvt(fma(s0,s1,s2))
// with default op_sel -> full-width f32 sources; x*1.0+0 == convert(x), RNE.
__device__ __forceinline__ unsigned pk_bf16(float a, float b) {
  float d;
  asm("v_fma_mixlo_bf16 %0, %1, 1.0, 0\n\t"
      "v_fma_mixhi_bf16 %0, %2, 1.0, 0"
      : "=&v"(d)
      : "v"(a), "v"(b));
  return __float_as_uint(d);
}

// A-matrix 16x32 bf16 layout (ISA 7.12.2): lane m=lane&15, half=lane>>4.
// dwords 0..3 hold K = half*8 + {0..7}; dwords 4..7 hold K = 16+half*8+{0..7}.
// rowptr points at this lane's A row, already offset by kk*32.
__device__ __forceinline__ void build_a(ABFrag& a, const float* rowptr, int half) {
  const float4 p0 = *(const float4*)(rowptr + half * 8);
  const float4 p1 = *(const float4*)(rowptr + half * 8 + 4);
  const float4 p2 = *(const float4*)(rowptr + 16 + half * 8);
  const float4 p3 = *(const float4*)(rowptr + 16 + half * 8 + 4);
  a.w[0] = pk_bf16(p0.x, p0.y); a.w[1] = pk_bf16(p0.z, p0.w);
  a.w[2] = pk_bf16(p1.x, p1.y); a.w[3] = pk_bf16(p1.z, p1.w);
  a.w[4] = pk_bf16(p2.x, p2.y); a.w[5] = pk_bf16(p2.z, p2.w);
  a.w[6] = pk_bf16(p3.x, p3.y); a.w[7] = pk_bf16(p3.z, p3.w);
}

__device__ __forceinline__ v8f wmma_bf16(v16bf a, v16bf b, v8f c) {
  return __builtin_amdgcn_wmma_f32_16x16x32_bf16(false, a, false, b,
                                                 (short)0, c, false, false);
}

// ---- pack W[K x NC] (row-major) into B-fragment images: per tile (kk,j),
// per lane, 8 consecutive dwords -> GEMMs load each frag as 2x b128. ----
__global__ __launch_bounds__(32) void k_pack_b(const float* __restrict__ W,
                                               unsigned* __restrict__ out,
                                               int Keff, int NC, int ntilesN) {
  const int lane = threadIdx.x, n = lane & 15, half = lane >> 4;
  const int kk = blockIdx.x / ntilesN, j = blockIdx.x % ntilesN;
  const int col = j * 16 + n;
  unsigned* o = out + ((size_t)blockIdx.x * 32 + lane) * 8;
#pragma unroll
  for (int d = 0; d < 8; ++d) {
    int kb = kk * 32 + (d < 4 ? half * 8 + 2 * d : 16 + half * 8 + 2 * (d - 4));     // B layout: K = half*8 + e (e<8), 16 + half*8 + (e-8) (e>=8)
    float f0 = (kb     < Keff) ? W[(size_t)kb * NC + col] : 0.0f;
    float f1 = (kb + 1 < Keff) ? W[(size_t)(kb + 1) * NC + col] : 0.0f;
    o[d] = pk_bf16(f0, f1);
  }
}

__device__ __forceinline__ void load_b(BFrag& b, const unsigned* packed,
                                       int tile, int lane) {
  const uint4* p = (const uint4*)(packed + ((size_t)tile * 32 + lane) * 8);
  b.q[0] = p[0]; b.q[1] = p[1];
}

// ---------------- [n,128] @ [128,64] + bias  (n % 16 == 0) ----------------
__global__ __launch_bounds__(32) void k_proj128(const float* __restrict__ A,
                                                const unsigned* __restrict__ pW,
                                                const float* __restrict__ bias,
                                                float* __restrict__ out, long n) {
  const int lane = threadIdx.x, m = lane & 15, half = lane >> 4;
  const long r0 = (long)blockIdx.x * 16;
  const int j = blockIdx.y;                       // 16-col tile of 64
  v8f acc = {};
#pragma unroll
  for (int kk = 0; kk < 4; ++kk) {                // K = 128
    ABFrag a; build_a(a, A + (r0 + m) * 128 + kk * 32, half);
    BFrag b; load_b(b, pW, kk * 4 + j, lane);
    acc = wmma_bf16(a.v, b.v, acc);
  }
  const float bc = bias[j * 16 + m];
#pragma unroll
  for (int r = 0; r < 8; ++r)
    out[(r0 + half * 8 + r) * 64 + j * 16 + m] = acc[r] + bc;
}

// ------------- [n,16] @ [16,64] + bias (K zero-padded to 32) -------------
// Producer of the 205MB `ea` stream: read input NT, write output NT so the
// stream does not thrash the 192MB L2 that caches the node tables.
__global__ __launch_bounds__(32) void k_proj16(const float* __restrict__ A,
                                               const unsigned* __restrict__ pW,
                                               const float* __restrict__ bias,
                                               float* __restrict__ out, long n) {
  const int lane = threadIdx.x, m = lane & 15, half = lane >> 4;
  const long r0 = (long)blockIdx.x * 16;
  const int j = blockIdx.y;
  ABFrag a;
  const f32x4 p0 = __builtin_nontemporal_load((const f32x4*)(A + (r0 + m) * 16 + half * 8));
  const f32x4 p1 = __builtin_nontemporal_load((const f32x4*)(A + (r0 + m) * 16 + half * 8 + 4));
  a.w[0] = pk_bf16(p0.x, p0.y); a.w[1] = pk_bf16(p0.z, p0.w);
  a.w[2] = pk_bf16(p1.x, p1.y); a.w[3] = pk_bf16(p1.z, p1.w);
  a.w[4] = 0u; a.w[5] = 0u; a.w[6] = 0u; a.w[7] = 0u;   // K = 16..31 padding
  BFrag b; load_b(b, pW, j, lane);                      // packed with zeros too
  v8f acc = {};
  acc = wmma_bf16(a.v, b.v, acc);
  const float bc = bias[j * 16 + m];
#pragma unroll
  for (int r = 0; r < 8; ++r)
    __builtin_nontemporal_store(acc[r] + bc,
        out + (r0 + half * 8 + r) * 64 + j * 16 + m);
}

// ---------------- z = relu(LayerNorm64(h)) ; one wave per node ----------------
__global__ __launch_bounds__(256) void k_ln_relu(const float* __restrict__ h,
                                                 const float* __restrict__ g,
                                                 const float* __restrict__ b,
                                                 float* __restrict__ z, long n) {
  const int lane = threadIdx.x & 31;
  const long node = (long)blockIdx.x * 8 + (threadIdx.x >> 5);
  if (node >= n) return;
  float v0 = h[node * 64 + lane], v1 = h[node * 64 + 32 + lane];
  float s = v0 + v1, ss = v0 * v0 + v1 * v1;
#pragma unroll
  for (int o = 16; o >= 1; o >>= 1) { s += __shfl_xor(s, o, 32); ss += __shfl_xor(ss, o, 32); }
  float mu = s * (1.0f / 64.0f);
  float var = ss * (1.0f / 64.0f) - mu * mu;
  float rs = rsqrtf(var + 1e-5f);
  float z0 = (v0 - mu) * rs * g[lane] + b[lane];
  float z1 = (v1 - mu) * rs * g[32 + lane] + b[32 + lane];
  z[node * 64 + lane]      = fmaxf(z0, 0.0f);
  z[node * 64 + 32 + lane] = fmaxf(z1, 0.0f);
}

// ---------------- edge pass 1: per-(dst,feat) max of positive logits ----------
// logits = (relu(z[src]+ea)+eps)*t > 0 with t>=0, so raw IEEE bits of positive
// f32 compare like u32 and memset(0) init reproduces segment_max->0 for empty.
// z gathers stay RT (L2-resident node table); ea stream is non-temporal.
__global__ __launch_bounds__(256) void k_edge_max(const float* __restrict__ z,
                                                  const float* __restrict__ ea,
                                                  const int* __restrict__ src,
                                                  const int* __restrict__ dst,
                                                  const float* __restrict__ tp,
                                                  unsigned* __restrict__ mx, long E) {
  long tid = (long)blockIdx.x * blockDim.x + threadIdx.x;
  if (tid >= E * 16) return;
  const float t = *tp;
  long e = tid >> 4; int q = (int)(tid & 15) * 4;
  long s = src[e], d = dst[e];
  float4 zv = *(const float4*)(z + s * 64 + q);
  f32x4 ev = __builtin_nontemporal_load((const f32x4*)(ea + e * 64 + q));
  float m0 = fmaxf(zv.x + ev.x, 0.0f) + MSG_EPS;
  float m1 = fmaxf(zv.y + ev.y, 0.0f) + MSG_EPS;
  float m2 = fmaxf(zv.z + ev.z, 0.0f) + MSG_EPS;
  float m3 = fmaxf(zv.w + ev.w, 0.0f) + MSG_EPS;
  unsigned* p = mx + d * 64 + q;
  atomicMax(p + 0, __float_as_uint(m0 * t));
  atomicMax(p + 1, __float_as_uint(m1 * t));
  atomicMax(p + 2, __float_as_uint(m2 * t));
  atomicMax(p + 3, __float_as_uint(m3 * t));
}

// ---------------- edge pass 2: num += msg*exp(l-m), den += exp(l-m) ----------
__global__ __launch_bounds__(256) void k_edge_acc(const float* __restrict__ z,
                                                  const float* __restrict__ ea,
                                                  const int* __restrict__ src,
                                                  const int* __restrict__ dst,
                                                  const float* __restrict__ tp,
                                                  const unsigned* __restrict__ mx,
                                                  float* __restrict__ num,
                                                  float* __restrict__ den, long E) {
  long tid = (long)blockIdx.x * blockDim.x + threadIdx.x;
  if (tid >= E * 16) return;
  const float t = *tp;
  long e = tid >> 4; int q = (int)(tid & 15) * 4;
  long s = src[e], d = dst[e];
  float4 zv = *(const float4*)(z + s * 64 + q);
  f32x4 ev = __builtin_nontemporal_load((const f32x4*)(ea + e * 64 + q));
  float m0 = fmaxf(zv.x + ev.x, 0.0f) + MSG_EPS;
  float m1 = fmaxf(zv.y + ev.y, 0.0f) + MSG_EPS;
  float m2 = fmaxf(zv.z + ev.z, 0.0f) + MSG_EPS;
  float m3 = fmaxf(zv.w + ev.w, 0.0f) + MSG_EPS;
  const unsigned* mp = mx + d * 64 + q;
  float w0 = __expf(m0 * t - __uint_as_float(mp[0]));
  float w1 = __expf(m1 * t - __uint_as_float(mp[1]));
  float w2 = __expf(m2 * t - __uint_as_float(mp[2]));
  float w3 = __expf(m3 * t - __uint_as_float(mp[3]));
  float* np = num + d * 64 + q;
  float* dp = den + d * 64 + q;
  atomicAdd(np + 0, m0 * w0); atomicAdd(dp + 0, w0);
  atomicAdd(np + 1, m1 * w1); atomicAdd(dp + 1, w1);
  atomicAdd(np + 2, m2 * w2); atomicAdd(dp + 2, w2);
  atomicAdd(np + 3, m3 * w3); atomicAdd(dp + 3, w3);
}

// ---- fused: u=num/den+x ; t1=relu(LN128(u@W1+b1)) ; out=[hres+] t1@W2+b2 ----
__global__ __launch_bounds__(32) void k_mlp(const float* __restrict__ xin,
                                            const float* __restrict__ num,
                                            const float* __restrict__ den,
                                            const unsigned* __restrict__ pW1,
                                            const float* __restrict__ b1,
                                            const float* __restrict__ g1,
                                            const float* __restrict__ be1,
                                            const unsigned* __restrict__ pW2,
                                            const float* __restrict__ b2,
                                            const float* __restrict__ hres,
                                            float* __restrict__ hout, long n) {
  __shared__ float uS[16][68];     // stride%64==4 -> conflict-free column reads
  __shared__ float t1S[16][132];
  const int lane = threadIdx.x, m = lane & 15, half = lane >> 4;
  const long r0 = (long)blockIdx.x * 16;

  // phase A: u = num/(den+1e-16) + xin  -> LDS (float4 granularity)
  for (int i = lane; i < 16 * 16; i += 32) {
    int r = i >> 4, c4 = (i & 15) * 4;
    long off = (r0 + r) * 64 + c4;
    float4 nv = *(const float4*)(num + off);
    float4 dv = *(const float4*)(den + off);
    float4 xv = *(const float4*)(xin + off);
    float4 u;
    u.x = nv.x / (dv.x + 1e-16f) + xv.x;
    u.y = nv.y / (dv.y + 1e-16f) + xv.y;
    u.z = nv.z / (dv.z + 1e-16f) + xv.z;
    u.w = nv.w / (dv.w + 1e-16f) + xv.w;
    *(float4*)&uS[r][c4] = u;
  }
  __syncthreads();

  // GEMM1: [16,64] @ [64,128]
  ABFrag a1[2];
  build_a(a1[0], &uS[m][0], half);
  build_a(a1[1], &uS[m][32], half);
#pragma unroll
  for (int j = 0; j < 8; ++j) {
    v8f acc = {};
#pragma unroll
    for (int kk = 0; kk < 2; ++kk) {
      BFrag b; load_b(b, pW1, kk * 8 + j, lane);
      acc = wmma_bf16(a1[kk].v, b.v, acc);
    }
    const float bc = b1[j * 16 + m];
#pragma unroll
    for (int r = 0; r < 8; ++r)
      t1S[half * 8 + r][j * 16 + m] = acc[r] + bc;
  }
  __syncthreads();

  // LayerNorm(128) + relu + scale/shift; 2 lanes per row
  {
    int row = lane >> 1, part = lane & 1;
    float s = 0.0f, ss = 0.0f;
    for (int c = part * 64; c < part * 64 + 64; ++c) {
      float v = t1S[row][c]; s += v; ss += v * v;
    }
    s += __shfl_xor(s, 1, 32);
    ss += __shfl_xor(ss, 1, 32);
    float mu = s * (1.0f / 128.0f);
    float var = ss * (1.0f / 128.0f) - mu * mu;
    float rs = rsqrtf(var + 1e-5f);
    for (int c = part * 64; c < part * 64 + 64; ++c) {
      float v = (t1S[row][c] - mu) * rs * g1[c] + be1[c];
      t1S[row][c] = fmaxf(v, 0.0f);
    }
  }
  __syncthreads();

  // GEMM2: [16,128] @ [128,64]; keep all 4 output tiles, store once
  ABFrag a2[4];
#pragma unroll
  for (int kk = 0; kk < 4; ++kk) build_a(a2[kk], &t1S[m][kk * 32], half);
  v8f acc2[4];
#pragma unroll
  for (int j = 0; j < 4; ++j) {
    v8f acc = {};
#pragma unroll
    for (int kk = 0; kk < 4; ++kk) {
      BFrag b; load_b(b, pW2, kk * 4 + j, lane);
      acc = wmma_bf16(a2[kk].v, b.v, acc);
    }
    acc2[j] = acc;
  }

  if (hres) {
#pragma unroll
    for (int j = 0; j < 4; ++j) {
      const float bc = b2[j * 16 + m];
#pragma unroll
      for (int r = 0; r < 8; ++r) {
        long off = (r0 + half * 8 + r) * 64 + j * 16 + m;
        hout[off] = acc2[j][r] + bc + hres[off];
      }
    }
  } else {
#pragma unroll
    for (int j = 0; j < 4; ++j) {
      const float bc = b2[j * 16 + m];
#pragma unroll
      for (int r = 0; r < 8; ++r)
        hout[(r0 + half * 8 + r) * 64 + j * 16 + m] = acc2[j][r] + bc;
    }
  }
}

// ---------------- head: out = relu(LN64(h)) @ lin_W + lin_b ----------------
__global__ __launch_bounds__(256) void k_head(const float* __restrict__ h,
                                              const float* __restrict__ g,
                                              const float* __restrict__ b,
                                              const float* __restrict__ W,
                                              const float* __restrict__ bias,
                                              float* __restrict__ out, long n) {
  const int lane = threadIdx.x & 31;
  const long node = (long)blockIdx.x * 8 + (threadIdx.x >> 5);
  if (node >= n) return;
  float v0 = h[node * 64 + lane], v1 = h[node * 64 + 32 + lane];
  float s = v0 + v1, ss = v0 * v0 + v1 * v1;
#pragma unroll
  for (int o = 16; o >= 1; o >>= 1) { s += __shfl_xor(s, o, 32); ss += __shfl_xor(ss, o, 32); }
  float mu = s * (1.0f / 64.0f);
  float var = ss * (1.0f / 64.0f) - mu * mu;
  float rs = rsqrtf(var + 1e-5f);
  float z0 = fmaxf((v0 - mu) * rs * g[lane] + b[lane], 0.0f);
  float z1 = fmaxf((v1 - mu) * rs * g[32 + lane] + b[32 + lane], 0.0f);
#pragma unroll
  for (int o = 0; o < 8; ++o) {
    float p = z0 * W[lane * 8 + o] + z1 * W[(32 + lane) * 8 + o];
#pragma unroll
    for (int sh = 16; sh >= 1; sh >>= 1) p += __shfl_xor(p, sh, 32);
    if (lane == 0) out[node * 8 + o] = p + bias[o];
  }
}

// ---------------------------------------------------------------------------
extern "C" void kernel_launch(void* const* d_in, const int* in_sizes, int n_in,
                              void* d_out, int out_size, void* d_ws, size_t ws_size,
                              hipStream_t stream) {
  const float* x         = (const float*)d_in[0];
  const float* edge_attr = (const float*)d_in[1];
  const float* node_W    = (const float*)d_in[2];
  const float* node_b    = (const float*)d_in[3];
  const float* edge_W    = (const float*)d_in[4];
  const float* edge_b    = (const float*)d_in[5];
  const float* conv_t    = (const float*)d_in[6];
  const float* conv_W1   = (const float*)d_in[7];
  const float* conv_b1   = (const float*)d_in[8];
  const float* conv_g1   = (const float*)d_in[9];
  const float* conv_be1  = (const float*)d_in[10];
  const float* conv_W2   = (const float*)d_in[11];
  const float* conv_b2   = (const float*)d_in[12];
  const float* ln_g      = (const float*)d_in[13];
  const float* ln_b      = (const float*)d_in[14];
  const float* lin_W     = (const float*)d_in[15];
  const float* lin_b     = (const float*)d_in[16];
  const int*   ei        = (const int*)d_in[17];   // per harness contract: int32

  const long N = in_sizes[0] / 128;   // 50000, multiple of 16
  const long E = in_sizes[1] / 16;    // 800000, multiple of 16

  // workspace: h | z | ea | mx | num | den | packed weights  (~269 MB)
  float* h   = (float*)d_ws;
  float* z   = h + N * 64;
  float* ea  = z + N * 64;
  unsigned* mx = (unsigned*)(ea + E * 64);
  float* num = (float*)mx + N * 64;
  float* den = num + N * 64;
  unsigned* pNW = (unsigned*)(den + N * 64);  // 16 tiles * 256 dwords
  unsigned* pEW = pNW + 16 * 256;             //  4 tiles
  unsigned* pW1 = pEW + 4 * 256;              // 4 layers * 16 tiles
  unsigned* pW2 = pW1 + 4 * 16 * 256;         // 4 layers * 16 tiles

  const dim3 w32(32);

  // pre-pack all weight matrices into WMMA B-fragment images
  k_pack_b<<<16, w32, 0, stream>>>(node_W, pNW, 128, 64, 4);
  k_pack_b<<<4,  w32, 0, stream>>>(edge_W, pEW, 16, 64, 4);   // K zero-padded
  for (int l = 0; l < 4; ++l) {
    k_pack_b<<<16, w32, 0, stream>>>(conv_W1 + (long)l * 64 * 128, pW1 + l * 4096, 64, 128, 8);
    k_pack_b<<<16, w32, 0, stream>>>(conv_W2 + (long)l * 128 * 64, pW2 + l * 4096, 128, 64, 4);
  }

  // input projections (WMMA)
  k_proj128<<<dim3((unsigned)(N / 16), 4), w32, 0, stream>>>(x, pNW, node_b, h, N);
  k_proj16 <<<dim3((unsigned)(E / 16), 4), w32, 0, stream>>>(edge_attr, pEW, edge_b, ea, E);

  const int edgeBlocks = (int)((E * 16 + 255) / 256);

  for (int l = 0; l < 4; ++l) {
    const float* feat = (l == 0) ? h : z;
    if (l > 0)
      k_ln_relu<<<(unsigned)((N + 7) / 8), 256, 0, stream>>>(h, ln_g + l * 64, ln_b + l * 64, z, N);
    hipMemsetAsync(mx, 0, (size_t)N * 64 * sizeof(float) * 3, stream);  // mx|num|den
    k_edge_max<<<edgeBlocks, 256, 0, stream>>>(feat, ea, ei, ei + E, conv_t + l, mx, E);
    k_edge_acc<<<edgeBlocks, 256, 0, stream>>>(feat, ea, ei, ei + E, conv_t + l, mx, num, den, E);
    k_mlp<<<(unsigned)(N / 16), w32, 0, stream>>>(
        feat, num, den,
        pW1 + l * 4096, conv_b1 + l * 128, conv_g1 + l * 128, conv_be1 + l * 128,
        pW2 + l * 4096, conv_b2 + l * 64,
        (l == 0) ? (const float*)nullptr : h, h, N);
  }

  k_head<<<(unsigned)((N + 7) / 8), 256, 0, stream>>>(h, ln_g, ln_b, lin_W, lin_b,
                                                      (float*)d_out, N);
}
